// DeepSetsClassifier_21182778704767
// MI455X (gfx1250) — matrix operand order repaired and checked by reading, not compiled, not run on hardware
//
#include <hip/hip_runtime.h>
#include <hip/hip_bf16.h>
#include <math.h>

#define IN_DIM 64
#define HID    128
#define LAT    128
#define NSEG   4096
#define MT     64      // points per tile (4 waves x 16 rows)
#define NBLK   2048    // persistent blocks (grid-stride over tiles)

// LDS strides (elements): padded to break 64-bank conflicts, rows 16B aligned
#define XS   72        // Xh  row stride (f16)
#define W1S  72        // W1t row stride (f16)  -> 128*72 halves, /8 = 1152 uint4
#define HS   136       // Hs  row stride (f16)
#define W2S  136       // W2t row stride (f16)  -> 128*136 halves, /8 = 2176 uint4
#define ES   132       // Enc row stride (f32)

typedef __attribute__((ext_vector_type(16))) _Float16 v16h;
typedef __attribute__((ext_vector_type(8)))  _Float16 v8h;
typedef __attribute__((ext_vector_type(8)))  float    v8f;

// A-operand loader for V_WMMA_*_16X16X32_F16 (16-bit A 16x32, contiguous k order used by this code, B loaded the same way):
// lane<16: M=lane, K={k0..k0+15}; lane>=16: same M, K={k0+16..k0+31}.
// caller passes p = row_base + k0 + (lane>>4)*16 ; two contiguous 16B chunks.
__device__ __forceinline__ v16h load_a(const _Float16* p) {
  v8h lo = *(const v8h*)(p);
  v8h hi = *(const v8h*)(p + 8);
  v16h a;
#pragma unroll
  for (int i = 0; i < 8; ++i) { a[i] = lo[i]; a[i + 8] = hi[i]; }
  return a;
}

// single-instruction ReLU: v_med3_f32(x, 0, +huge)
__device__ __forceinline__ float relu1(float x) {
  return __builtin_amdgcn_fmed3f(x, 0.f, 3.0e38f);
}

// order-preserving float<->uint key (atomicMax over signed floats)
__device__ __forceinline__ unsigned fkey(float f) {
  unsigned u = __float_as_uint(f);
  return (u & 0x80000000u) ? ~u : (u | 0x80000000u);
}
__device__ __forceinline__ float funkey(unsigned k) {
  unsigned u = (k & 0x80000000u) ? (k & 0x7fffffffu) : ~k;
  return __uint_as_float(u);
}

__global__ __launch_bounds__(256)
void init_accum_kernel(float* __restrict__ gsum, unsigned* __restrict__ gmaxk,
                       unsigned* __restrict__ gcnt, int n) {
  int i = blockIdx.x * blockDim.x + threadIdx.x;
  if (i < n) { gsum[i] = 0.f; gmaxk[i] = 0u; }
  if (i < NSEG) gcnt[i] = 0u;
}

// One-shot: convert weights to f16, transposed [n][k], padded strides (B layout)
__global__ __launch_bounds__(256)
void conv_weights_kernel(const float* __restrict__ w1, const float* __restrict__ w2,
                         _Float16* __restrict__ w1h, _Float16* __restrict__ w2h) {
  int i = blockIdx.x * blockDim.x + threadIdx.x;
  if (i < HID * W1S) {
    int n = i / W1S, k = i % W1S;
    w1h[i] = (k < IN_DIM) ? (_Float16)w1[k * HID + n] : (_Float16)0.f;
  }
  if (i < LAT * W2S) {
    int n = i / W2S, k = i % W2S;
    w2h[i] = (k < HID) ? (_Float16)w2[k * LAT + n] : (_Float16)0.f;
  }
}

// Persistent fused kernel: stage weights once, then grid-stride over point tiles:
// phi MLP (f16 WMMA, f32 accum) + sorted-segment tile scan + global atomic flush.
__global__ __launch_bounds__(128)
void phi_pool_kernel(const float* __restrict__ x, const int* __restrict__ bidx,
                     const _Float16* __restrict__ w1h, const _Float16* __restrict__ w2h,
                     const float* __restrict__ b1, const float* __restrict__ b2,
                     float* __restrict__ gsum, unsigned* __restrict__ gmaxk,
                     unsigned* __restrict__ gcnt, int npts, int ntiles) {
  __shared__ __align__(16) _Float16 Xh [MT * XS];    // x tile, f16 [m][k]
  __shared__ __align__(16) _Float16 W1t[HID * W1S];  // W1^T [n][k], f16
  __shared__ __align__(16) _Float16 Hsh[MT * HS];    // ReLU hidden, f16 [m][h]
  __shared__ __align__(16) _Float16 W2t[LAT * W2S];  // W2^T [n][k], f16
  __shared__ float    Enc[MT * ES];                  // f32 encodings [m][n]
  __shared__ unsigned segs[MT];
  __shared__ float    b1s[HID];
  __shared__ float    b2s[LAT];

  const int tid = threadIdx.x;

  // ---- stage pre-converted weights once per block (raw 16B copies) ----
  {
    const uint4* s1 = (const uint4*)w1h; uint4* d1 = (uint4*)W1t;
    for (int i = tid; i < (HID * W1S) / 8; i += 128) d1[i] = s1[i];
    const uint4* s2 = (const uint4*)w2h; uint4* d2 = (uint4*)W2t;
    for (int i = tid; i < (LAT * W2S) / 8; i += 128) d2[i] = s2[i];
    if (tid < HID) b1s[tid] = b1[tid];
    if (tid < LAT) b2s[tid] = b2[tid];
  }

  const int lane = tid & 31;
  const int wv   = tid >> 5;       // 4 waves
  const int hl   = lane >> 4;      // lane half (K select)
  const int rr   = lane & 15;      // A row / B column within 16x16 tile
  const int m0   = wv * 16;        // wave's 16 rows
  const _Float16* arow = &Xh [(m0 + rr) * XS];
  const _Float16* hrow = &Hsh[(m0 + rr) * HS];

  for (int tile = blockIdx.x; tile < ntiles; tile += gridDim.x) {
    const int m_base = tile * MT;

    // prefetch next tile's x (16KB, 128B per thread) while this tile computes
    {
      int nb = m_base + (int)gridDim.x * MT;
      if (nb < npts) {
        const char* p = (const char*)(x + (size_t)nb * IN_DIM);
        __builtin_prefetch(p + tid * 128, 0, 1);
      }
    }

    __syncthreads();   // previous iteration's scan (Enc/segs) done before restage

    // ---- stage x tile (f32 -> f16) + segment ids ----
    for (int i = tid; i < MT * IN_DIM; i += 128) {
      int k = i & (IN_DIM - 1), m = i >> 6;
      int g = m_base + m;
      Xh[m * XS + k] = (g < npts) ? (_Float16)x[(size_t)g * IN_DIM + k] : (_Float16)0.f;
    }
    if (tid < MT) {
      int g = m_base + tid;
      segs[tid] = (g < npts) ? (unsigned)bidx[g] : 0xFFFFFFFFu;
    }
    __syncthreads();

    // ---- GEMM1: H = relu(X @ W1 + b1); A fragments hoisted across n-tiles ----
    {
      v16h a0 = load_a(arow + 0  + hl * 16);
      v16h a1 = load_a(arow + 32 + hl * 16);
#pragma unroll
      for (int nt = 0; nt < HID / 16; ++nt) {
        float bias = b1s[nt * 16 + rr];
        v8f acc;
#pragma unroll
        for (int r = 0; r < 8; ++r) acc[r] = bias;
        v16h bm0 = *(const v16h*)&W1t[(nt * 16 + rr) * W1S + 0  + hl * 16];
        v16h bm1 = *(const v16h*)&W1t[(nt * 16 + rr) * W1S + 32 + hl * 16];
        acc = __builtin_amdgcn_wmma_f32_16x16x32_f16(false, a0, false, bm0,
                                                     (short)0, acc, false, false);
        acc = __builtin_amdgcn_wmma_f32_16x16x32_f16(false, a1, false, bm1,
                                                     (short)0, acc, false, false);
#pragma unroll
        for (int r = 0; r < 8; ++r)
          Hsh[(m0 + r + 8 * hl) * HS + nt * 16 + rr] = (_Float16)relu1(acc[r]);
      }
    }
    __syncthreads();

    // ---- GEMM2: Enc = H @ W2 + b2; A fragments hoisted across n-tiles ----
    {
      v16h ha[4];
#pragma unroll
      for (int ks = 0; ks < 4; ++ks) ha[ks] = load_a(hrow + ks * 32 + hl * 16);
#pragma unroll
      for (int nt = 0; nt < LAT / 16; ++nt) {
        float bias = b2s[nt * 16 + rr];
        v8f acc;
#pragma unroll
        for (int r = 0; r < 8; ++r) acc[r] = bias;
#pragma unroll
        for (int ks = 0; ks < 4; ++ks) {
          v16h bm = *(const v16h*)&W2t[(nt * 16 + rr) * W2S + ks * 32 + hl * 16];
          acc = __builtin_amdgcn_wmma_f32_16x16x32_f16(false, ha[ks], false, bm,
                                                       (short)0, acc, false, false);
        }
#pragma unroll
        for (int r = 0; r < 8; ++r)
          Enc[(m0 + r + 8 * hl) * ES + nt * 16 + rr] = acc[r];
      }
    }
    __syncthreads();

    // ---- tile-local segment scan (sorted index => few runs), atomic flush ----
    {
      const int col = tid;  // 128 threads == LAT columns
      unsigned cur = 0xFFFFFFFFu, rl = 0;
      float s = 0.f, mx = -3.402823466e38f;
#pragma unroll 1
      for (int r = 0; r < MT; ++r) {
        unsigned sg = segs[r];
        if (sg == 0xFFFFFFFFu) continue;   // padded tail rows
        if (sg != cur) {
          if (cur != 0xFFFFFFFFu) {
            atomicAdd(&gsum[(size_t)cur * LAT + col], s);
            atomicMax(&gmaxk[(size_t)cur * LAT + col], fkey(mx));
            if (col == 0) atomicAdd(&gcnt[cur], rl);
          }
          cur = sg; s = 0.f; mx = -3.402823466e38f; rl = 0;
        }
        float v = Enc[r * ES + col];
        s += v; mx = fmaxf(mx, v); ++rl;
      }
      if (cur != 0xFFFFFFFFu) {
        atomicAdd(&gsum[(size_t)cur * LAT + col], s);
        atomicMax(&gmaxk[(size_t)cur * LAT + col], fkey(mx));
        if (col == 0) atomicAdd(&gcnt[cur], rl);
      }
    }
  }
}

// Finalize pooling + rho MLP: one block (128 threads) per segment
__global__ __launch_bounds__(128)
void rho_kernel(const float* __restrict__ gsum, const unsigned* __restrict__ gmaxk,
                const unsigned* __restrict__ gcnt,
                const float* __restrict__ rw1, const float* __restrict__ rb1,
                const float* __restrict__ rw2, const float* __restrict__ rb2,
                float* __restrict__ out) {
  __shared__ float pooled[2 * LAT + 1];
  __shared__ float red[128];
  const int seg = blockIdx.x;
  const int tid = threadIdx.x;

  unsigned cnt = gcnt[seg];
  float cc = fmaxf((float)cnt, 1.0f);
  pooled[tid] = gsum[(size_t)seg * LAT + tid] / cc;                    // mean
  pooled[LAT + tid] = (cnt > 0u) ? funkey(gmaxk[(size_t)seg * LAT + tid])
                                 : -3.402823466e38f;                   // max (finfo.min if empty)
  if (tid == 0) pooled[2 * LAT] = logf(cc);                            // log count
  __syncthreads();

  float h = rb1[tid];
#pragma unroll 1
  for (int k = 0; k < 2 * LAT + 1; ++k)
    h = fmaf(pooled[k], rw1[k * HID + tid], h);   // coalesced across tid
  h = relu1(h);
  red[tid] = h * rw2[tid];
  __syncthreads();
  for (int s = 64; s > 0; s >>= 1) {
    if (tid < s) red[tid] += red[tid + s];
    __syncthreads();
  }
  if (tid == 0) out[seg] = red[0] + rb2[0];
}

extern "C" void kernel_launch(void* const* d_in, const int* in_sizes, int n_in,
                              void* d_out, int out_size, void* d_ws, size_t ws_size,
                              hipStream_t stream) {
  (void)n_in; (void)out_size; (void)ws_size;
  const float* x   = (const float*)d_in[0];
  const int*   bi  = (const int*)d_in[1];
  const float* w1  = (const float*)d_in[2];
  const float* b1  = (const float*)d_in[3];
  const float* w2  = (const float*)d_in[4];
  const float* b2  = (const float*)d_in[5];
  const float* rw1 = (const float*)d_in[6];
  const float* rb1 = (const float*)d_in[7];
  const float* rw2 = (const float*)d_in[8];
  const float* rb2 = (const float*)d_in[9];
  float* out = (float*)d_out;

  const int npts   = in_sizes[0] / IN_DIM;
  const int ntiles = (npts + MT - 1) / MT;

  // workspace layout (16B-aligned chunks)
  char* ws = (char*)d_ws;
  float*    gsum  = (float*)ws;                                   // 4096*128*4 = 2MB
  unsigned* gmaxk = (unsigned*)(ws + (size_t)NSEG * LAT * 4);     // 2MB
  unsigned* gcnt  = (unsigned*)(ws + (size_t)NSEG * LAT * 8);     // 16KB
  _Float16* w1h   = (_Float16*)(ws + (size_t)NSEG * LAT * 8 + NSEG * 4);
  _Float16* w2h   = w1h + HID * W1S;

  const int accN = NSEG * LAT;
  hipLaunchKernelGGL(init_accum_kernel, dim3((accN + 255) / 256), dim3(256), 0, stream,
                     gsum, gmaxk, gcnt, accN);
  hipLaunchKernelGGL(conv_weights_kernel, dim3((LAT * W2S + 255) / 256), dim3(256), 0, stream,
                     w1, w2, w1h, w2h);
  int nblk = ntiles < NBLK ? ntiles : NBLK;
  hipLaunchKernelGGL(phi_pool_kernel, dim3(nblk), dim3(128), 0, stream,
                     x, bi, w1h, w2h, b1, b2, gsum, gmaxk, gcnt, npts, ntiles);
  hipLaunchKernelGGL(rho_kernel, dim3(NSEG), dim3(128), 0, stream,
                     gsum, gmaxk, gcnt, rw1, rb1, rw2, rb2, out);
}
